// NakaCorrectionLossWithMasks_76605036692173
// MI455X (gfx1250) — compile-verified, matrix operand order checked by reading, not run on hardware
//
#include <hip/hip_runtime.h>

typedef __attribute__((ext_vector_type(16))) _Float16 v16h;
typedef __attribute__((ext_vector_type(8)))  float    v8f;

#define BATCH 16
#define H 512
#define W 512
#define SPIX (H*W)              // 262144 pixels per sample
#define NELEM (BATCH*3*SPIX)    // 12582912 elements per tensor

// workspace layout (32-bit word offsets)
#define OFF_HIST1   0u          // 16 * 16384 coarse bins
#define OFF_HIST2A  262144u     // 16 * 65536 fine bins (rank k)
#define OFF_HIST2B  1310720u    // 16 * 65536 fine bins (rank k+1)
#define OFF_LAPMAX  2359296u    // 16 (float bits)
#define OFF_SEL     2359312u    // 16 * {p0,c0,p1,c1}
#define OFF_ACC     2359376u    // 1 float accumulator
#define OFF_THR     2359392u    // 16 floats
#define WS_WORDS    2359424u

// rank targets: pos = 0.85*(SPIX-1) = 222821.55 -> k = 222821
#define RANK_T0 222822u   // (k+1)-th smallest as a 1-based count
#define RANK_T1 222823u

__device__ __forceinline__ int refl101(int i, int n) {
  if (i < 0) i = -i;
  if (i >= n) i = 2*n - 2 - i;
  return i;
}

__device__ __forceinline__ float gray_at(const float* img, int b, int y, int x) {
  const float* p = img + (size_t)b*3*SPIX + (size_t)y*W + x;
  return 0.299f*p[0] + 0.587f*p[SPIX] + 0.114f*p[2*SPIX];
}

// |2*(corners) - 8*center| with BORDER_REFLECT_101
__device__ __forceinline__ float lap_abs_at(const float* gt, int b, int y, int x) {
  int ym = refl101(y-1, H), yp = refl101(y+1, H);
  int xm = refl101(x-1, W), xp = refl101(x+1, W);
  float c = gray_at(gt, b, y, x);
  float s = gray_at(gt, b, ym, xm) + gray_at(gt, b, ym, xp)
          + gray_at(gt, b, yp, xm) + gray_at(gt, b, yp, xp);
  return fabsf(2.0f*s - 8.0f*c);
}

// ---------------- K0: zero workspace ----------------
__global__ void naka_k0_zero(unsigned* ws, unsigned n) {
  for (unsigned i = blockIdx.x*blockDim.x + threadIdx.x; i < n; i += gridDim.x*blockDim.x)
    ws[i] = 0u;
}

// ---------------- K1: coarse histogram of gray(pred) + per-sample laplacian max ----------------
// grid 1024x256: thread t handles pixel t of every sample j (slot j <-> sample j)
__global__ void naka_k1_hist_lapmax(const float* __restrict__ pred,
                                    const float* __restrict__ gt,
                                    unsigned* __restrict__ hist1,
                                    unsigned* __restrict__ lapmax_bits) {
  int t = blockIdx.x*256 + threadIdx.x;       // 0..262143
  int lane = threadIdx.x & 31, wave = threadIdx.x >> 5;
  int y = t >> 9, x = t & 511;
  __shared__ float smax[16][8];

  float lm[16];
  #pragma unroll
  for (int j = 0; j < 16; ++j) {
    float gp = gray_at(pred, j, y, x);
    unsigned bits = __float_as_uint(gp);
    atomicAdd(&hist1[j*16384 + (bits >> 16)], 1u);
    if (j < 15) __builtin_prefetch(gt + (size_t)(j+1)*3*SPIX + t, 0, 1);  // global_prefetch_b8
    lm[j] = lap_abs_at(gt, j, y, x);
  }
  #pragma unroll
  for (int j = 0; j < 16; ++j) {
    float v = lm[j];
    v = fmaxf(v, __shfl_xor(v, 1));
    v = fmaxf(v, __shfl_xor(v, 2));
    v = fmaxf(v, __shfl_xor(v, 4));
    v = fmaxf(v, __shfl_xor(v, 8));
    v = fmaxf(v, __shfl_xor(v, 16));
    if (lane == 0) smax[j][wave] = v;
  }
  __syncthreads();
  if (threadIdx.x < 16) {
    float v = smax[threadIdx.x][0];
    #pragma unroll
    for (int w = 1; w < 8; ++w) v = fmaxf(v, smax[threadIdx.x][w]);
    atomicMax(&lapmax_bits[threadIdx.x], __float_as_uint(v));  // monotone for nonneg f32
  }
}

// ---------------- K2: select coarse bins containing ranks k and k+1 ----------------
__global__ void naka_k2_coarse_select(const unsigned* __restrict__ hist1,
                                      unsigned* __restrict__ sel) {
  int b = blockIdx.x, tid = threadIdx.x;   // 16 blocks x 256 threads
  const unsigned* h = hist1 + b*16384;
  __shared__ unsigned ssum[256];
  unsigned s = 0;
  for (int i = 0; i < 64; ++i) s += h[tid*64 + i];
  ssum[tid] = s; __syncthreads();
  for (int off = 1; off < 256; off <<= 1) {
    unsigned v = (tid >= off) ? ssum[tid-off] : 0u;
    __syncthreads();
    ssum[tid] += v;
    __syncthreads();
  }
  unsigned incl = ssum[tid], excl = tid ? ssum[tid-1] : 0u;
  #pragma unroll
  for (int which = 0; which < 2; ++which) {
    unsigned tt = which ? RANK_T1 : RANK_T0;
    if (excl < tt && tt <= incl) {
      unsigned c = excl;
      for (int i = 0; i < 64; ++i) {
        unsigned cnt = h[tid*64 + i];
        if (c + cnt >= tt) { sel[b*4 + which*2] = (unsigned)(tid*64 + i); sel[b*4 + which*2 + 1] = c; break; }
        c += cnt;
      }
    }
  }
}

// ---------------- K3: fine histograms for the two selected prefixes ----------------
__global__ void naka_k3_fine_hist(const float* __restrict__ pred,
                                  const unsigned* __restrict__ sel,
                                  unsigned* __restrict__ h2a,
                                  unsigned* __restrict__ h2b) {
  int t = blockIdx.x*256 + threadIdx.x;
  int y = t >> 9, x = t & 511;
  __shared__ unsigned sp[16][2];
  if (threadIdx.x < 32) sp[threadIdx.x >> 1][threadIdx.x & 1] = sel[(threadIdx.x >> 1)*4 + (threadIdx.x & 1)*2];
  __syncthreads();
  #pragma unroll
  for (int j = 0; j < 16; ++j) {
    unsigned bits = __float_as_uint(gray_at(pred, j, y, x));
    unsigned coarse = bits >> 16, fine = bits & 0xFFFFu;
    if (coarse == sp[j][0]) atomicAdd(&h2a[j*65536 + fine], 1u);
    if (coarse == sp[j][1]) atomicAdd(&h2b[j*65536 + fine], 1u);
  }
}

// ---------------- K3b: resolve order statistics -> interpolated threshold ----------------
__global__ void naka_k3b_threshold(const unsigned* __restrict__ h2a,
                                   const unsigned* __restrict__ h2b,
                                   const unsigned* __restrict__ sel,
                                   float* __restrict__ thr) {
  int b = blockIdx.x, tid = threadIdx.x;   // 16 blocks x 256 threads
  __shared__ unsigned ssum[256];
  __shared__ float sv[2];
  for (int which = 0; which < 2; ++which) {
    const unsigned* h = (which ? h2b : h2a) + b*65536;
    unsigned pcoarse = sel[b*4 + which*2];
    unsigned cbefore = sel[b*4 + which*2 + 1];
    unsigned tt = (which ? RANK_T1 : RANK_T0) - cbefore;   // rank within selected prefix bin
    unsigned s = 0;
    for (int i = 0; i < 256; ++i) s += h[tid*256 + i];
    ssum[tid] = s; __syncthreads();
    for (int off = 1; off < 256; off <<= 1) {
      unsigned v = (tid >= off) ? ssum[tid-off] : 0u;
      __syncthreads();
      ssum[tid] += v;
      __syncthreads();
    }
    unsigned incl = ssum[tid], excl = tid ? ssum[tid-1] : 0u;
    if (excl < tt && tt <= incl) {
      unsigned c = excl;
      for (int i = 0; i < 256; ++i) {
        unsigned cnt = h[tid*256 + i];
        if (c + cnt >= tt) { sv[which] = __uint_as_float((pcoarse << 16) | (unsigned)(tid*256 + i)); break; }
        c += cnt;
      }
    }
    __syncthreads();
  }
  if (tid == 0) {
    double pos = 0.85 * (double)(SPIX - 1);
    double frac = pos - floor(pos);     // ~0.55
    double v0 = (double)sv[0], v1 = (double)sv[1];
    thr[b] = (float)(v0 + frac*(v1 - v0));
  }
}

// ---------------- K4: fused weighted L1 reduction, WMMA as the reduction engine ----------------
__global__ void naka_k4_wmma_reduce(const float* __restrict__ pred,
                                    const float* __restrict__ gt,
                                    const unsigned* __restrict__ lapmax_bits,
                                    const float* __restrict__ thr,
                                    float* __restrict__ acc) {
  int t = blockIdx.x*256 + threadIdx.x;        // pixel t of each sample
  int lane = threadIdx.x & 31, wave = threadIdx.x >> 5;
  int y = t >> 9, x = t & 511;
  __shared__ unsigned sstage[32];              // [0:16)=thr bits, [16:32)=lapmax bits
  __shared__ float swsum[8];

  // Stage the 128B of per-sample constants into LDS with the gfx1250 async DMA
  // path (GLOBAL_LOAD_ASYNC_TO_LDS_B128, tracked by ASYNCcnt). 8 lanes of wave 0
  // each move 16 bytes; the issuing wave drains ASYNCcnt, the barrier publishes.
  if (threadIdx.x < 8) {
    const void* src = (threadIdx.x < 4)
        ? (const void*)(thr + threadIdx.x*4)
        : (const void*)(lapmax_bits + (threadIdx.x - 4)*4);
    unsigned lds_off = (unsigned)(size_t)(&sstage[threadIdx.x*4]);  // addrspace(3) byte offset
    unsigned long long gaddr = (unsigned long long)(size_t)src;
    asm volatile("global_load_async_to_lds_b128 %0, %1, off"
                 :: "v"(lds_off), "v"(gaddr) : "memory");
  }
  asm volatile("s_wait_asynccnt 0x0" ::: "memory");
  __syncthreads();

  v16h aones, bfrag;
  #pragma unroll
  for (int i = 0; i < 16; ++i) aones[i] = (_Float16)1.0f;

  #pragma unroll
  for (int j = 0; j < 16; ++j) {               // slot j == sample j
    const float* pp = pred + (size_t)j*3*SPIX + t;
    const float* gg = gt   + (size_t)j*3*SPIX + t;
    float ad = fabsf(pp[0] - gg[0]) + fabsf(pp[SPIX] - gg[SPIX]) + fabsf(pp[2*SPIX] - gg[2*SPIX]);
    float lmax = __uint_as_float(sstage[16 + j]) + 1e-8f;
    float gm = sqrtf(lap_abs_at(gt, j, y, x) / lmax);
    float bm = (gray_at(pred, j, y, x) >= __uint_as_float(sstage[j])) ? 1.0f : 0.0f;
    bfrag[j] = (_Float16)((1.0f + 0.5f*gm + 0.8f*bm) * ad);
  }

  // D[m,n] = sum_k B[k,n]; column sums (f32 accumulate). B holds this wave's 512 pixel values.
  v8f c = {};
  c = __builtin_amdgcn_wmma_f32_16x16x32_f16(false, aones, false, bfrag,
                                             (short)0, c, false, false);
  // lanes 0..15 hold colsum_0..15 in c[0]; sum them within each 16-lane half
  float s = c[0];
  s += __shfl_xor(s, 1);
  s += __shfl_xor(s, 2);
  s += __shfl_xor(s, 4);
  s += __shfl_xor(s, 8);
  if (lane == 0) swsum[wave] = s;              // wave total
  __syncthreads();
  if (threadIdx.x == 0) {
    float tot = 0.0f;
    #pragma unroll
    for (int w = 0; w < 8; ++w) tot += swsum[w];
    atomicAdd(acc, tot);
  }
}

// ---------------- K5: finalize ----------------
__global__ void naka_k5_finalize(const float* __restrict__ acc, float* __restrict__ out) {
  out[0] = (float)((double)acc[0] / (double)NELEM);
}

extern "C" void kernel_launch(void* const* d_in, const int* in_sizes, int n_in,
                              void* d_out, int out_size, void* d_ws, size_t ws_size,
                              hipStream_t stream) {
  const float* pred = (const float*)d_in[0];
  const float* gt   = (const float*)d_in[1];
  // d_in[2] ("naka") is unused by the reference loss
  float* out = (float*)d_out;

  unsigned* ws     = (unsigned*)d_ws;
  unsigned* hist1  = ws + OFF_HIST1;
  unsigned* h2a    = ws + OFF_HIST2A;
  unsigned* h2b    = ws + OFF_HIST2B;
  unsigned* lapmax = ws + OFF_LAPMAX;
  unsigned* sel    = ws + OFF_SEL;
  float*    acc    = (float*)(ws + OFF_ACC);
  float*    thr    = (float*)(ws + OFF_THR);

  naka_k0_zero<<<2048, 256, 0, stream>>>(ws, WS_WORDS);
  naka_k1_hist_lapmax<<<1024, 256, 0, stream>>>(pred, gt, hist1, lapmax);
  naka_k2_coarse_select<<<16, 256, 0, stream>>>(hist1, sel);
  naka_k3_fine_hist<<<1024, 256, 0, stream>>>(pred, sel, h2a, h2b);
  naka_k3b_threshold<<<16, 256, 0, stream>>>(h2a, h2b, sel, thr);
  naka_k4_wmma_reduce<<<1024, 256, 0, stream>>>(pred, gt, lapmax, thr, acc);
  naka_k5_finalize<<<1, 1, 0, stream>>>(acc, out);
}
